// Block_56401510531518
// MI455X (gfx1250) — matrix operand order repaired and checked by reading, not compiled, not run on hardware
//
#include <hip/hip_runtime.h>
#include <hip/hip_bf16.h>
#include <stdint.h>

// ---------------------------------------------------------------------------
// Transformer block (LN1 -> MHA(causal) -> +res -> LN2 -> FFN -> +res)
// B=2048, T=64, D=384, H=8, HS=48, F=1536.  All GEMMs on bf16 WMMA, f32 accum.
// A-panels staged into LDS via the Tensor Data Mover; weights pre-swizzled
// into WMMA-B fragment order so every B-fragment is one 32B per-lane load.
// ---------------------------------------------------------------------------

#define BATCH 2048
#define TT    64
#define DD    384
#define NH    8
#define HS    48
#define FF    1536

typedef __attribute__((ext_vector_type(16))) __bf16 v16bf;
typedef __attribute__((ext_vector_type(8)))  __bf16 v8bf;
typedef __attribute__((ext_vector_type(8)))  float  v8f;
typedef __attribute__((ext_vector_type(4)))  unsigned int u32x4;
typedef __attribute__((ext_vector_type(8)))  int i32x8;
typedef __attribute__((ext_vector_type(4)))  int i32x4;

__device__ __forceinline__ v8f zero8() {
  v8f z;
#pragma unroll
  for (int i = 0; i < 8; ++i) z[i] = 0.0f;
  return z;
}

__device__ __forceinline__ v8f wmma_bf16(v16bf a, v16bf b, v8f c) {
  return __builtin_amdgcn_wmma_f32_16x16x32_bf16(
      false, a, false, b, (short)0, c, false, false);
}

// ---------------------------------------------------------------------------
// Tensor Data Mover: DMA a 2D bf16 tile into contiguous LDS (D# per ISA ch.8).
// Issue from ONE wave; that wave waits TENSORcnt, then workgroup barrier.
// ---------------------------------------------------------------------------
__device__ __forceinline__ void tdm_load_2d_bf16(
    unsigned lds_off, const void* gptr, unsigned tensor_d0, unsigned tensor_d1,
    unsigned stride0, unsigned tile_d0, unsigned tile_d1) {
  unsigned long long ga = (unsigned long long)(uintptr_t)gptr;
  u32x4 g0;
  g0[0] = 1u;                                   // count=1 (valid user D#)
  g0[1] = lds_off;                              // lds_addr (bytes)
  g0[2] = (unsigned)(ga & 0xFFFFFFFFu);         // global_addr[31:0]
  g0[3] = (unsigned)((ga >> 32) & 0x01FFFFFFu)  // global_addr[56:32]
          | 0x80000000u;                        // type=2 ("image")
  i32x8 g1;
  g1[0] = (int)(1u << 16);                      // data_size=1 -> 2 bytes
  g1[1] = (int)((tensor_d0 & 0xFFFFu) << 16);   // tensor_dim0[15:0]
  g1[2] = (int)((tensor_d0 >> 16) | ((tensor_d1 & 0xFFFFu) << 16));
  g1[3] = (int)((tensor_d1 >> 16) | (tile_d0 << 16));   // tile_dim0
  g1[4] = (int)(tile_d1 & 0xFFFFu);             // tile_dim1 (tile_dim2=0)
  g1[5] = (int)stride0;                         // tensor_dim0_stride[31:0]
  g1[6] = 0;
  g1[7] = 0;
  i32x4 z4;
  z4[0] = 0; z4[1] = 0; z4[2] = 0; z4[3] = 0;
#if defined(__clang_major__) && (__clang_major__ >= 23)
  i32x8 z8;
#pragma unroll
  for (int i = 0; i < 8; ++i) z8[i] = 0;
  __builtin_amdgcn_tensor_load_to_lds(g0, g1, z4, z4, z8, 0);
#else
  __builtin_amdgcn_tensor_load_to_lds(g0, g1, z4, z4, 0);
#endif
}

// A-fragment (16x32 bf16) from LDS.  Contiguous K per lane half:
// k = hi*16 + e (elems 0-15), matching the B fragments,
// so this is exactly two 16-byte LDS loads.
__device__ __forceinline__ v16bf load_a_lds(const __bf16* base, int ld, int lane) {
  int m  = lane & 15;
  int hi = (lane >> 4) & 1;
  const __bf16* p = base + m * ld + hi * 16;
  v8bf lo = *(const v8bf*)p;
  v8bf h8 = *(const v8bf*)(p + 8);
  return __builtin_shufflevector(lo, h8, 0, 1, 2, 3, 4, 5, 6, 7,
                                 8, 9, 10, 11, 12, 13, 14, 15);
}

// B-fragment whose k index runs contiguously in memory per lane:
// element(k,n) = base[n*ld + k], k = hi*16 + e  ->  one 32B per-lane load.
__device__ __forceinline__ v16bf load_bT_lds(const __bf16* base, int ld, int lane) {
  int n  = lane & 15;
  int hi = (lane >> 4) & 1;
  return *(const v16bf*)(base + n * ld + hi * 16);
}

// B-fragment from pre-swizzled GLOBAL weights: frag-major, lane-major, 32B/lane.
__device__ __forceinline__ v16bf load_b_packed(const __bf16* pk, int frag, int lane) {
  return ((const v16bf*)pk)[frag * 32 + lane];
}

// ---------------------------------------------------------------------------
// Kernel 0: pack f32 [K x N] row-major weights into bf16 WMMA-B fragment
// order: dst[((kt*NT + nt)*32 + lane)*16 + e] = w[kt*32 + hi*16 + e][nt*16+n].
// blockIdx.y selects the matrix for batched (per-head) weights.
// ---------------------------------------------------------------------------
__global__ __launch_bounds__(256) void packB_kernel(
    const float* __restrict__ src0, __bf16* __restrict__ dst0, int K, int N) {
  const float* src = src0 + (size_t)blockIdx.y * K * N;
  __bf16* dst      = dst0 + (size_t)blockIdx.y * K * N;
  int t = blockIdx.x * blockDim.x + threadIdx.x;
  int lane = t & 31, frag = t >> 5;
  int NT = N >> 4;
  if (frag >= (K >> 5) * NT) return;
  int kt = frag / NT, nt = frag - kt * NT;
  int n = lane & 15, hi = (lane >> 4) & 1;
  const float* s = src + (size_t)(kt * 32 + hi * 16) * N + nt * 16 + n;
  v16bf v;
#pragma unroll
  for (int e = 0; e < 16; ++e) v[e] = (__bf16)s[(size_t)e * N];
  *(v16bf*)(dst + ((size_t)frag * 32 + lane) * 16) = v;
}

// ---------------------------------------------------------------------------
// Kernel 1: LayerNorm(x) -> hnorm f32 + bf16.  One wave per row.
// ---------------------------------------------------------------------------
__global__ __launch_bounds__(256) void ln1_kernel(
    const float* __restrict__ x, const float* __restrict__ g,
    const float* __restrict__ be, float* __restrict__ hn,
    __bf16* __restrict__ hnb, int nrows) {
  int row  = blockIdx.x * (blockDim.x >> 5) + (threadIdx.x >> 5);
  int lane = threadIdx.x & 31;
  if (row >= nrows) return;
  const float* xr = x + (size_t)row * DD;
  float v[12];
  float s = 0.0f;
#pragma unroll
  for (int i = 0; i < 12; ++i) { v[i] = xr[lane + i * 32]; s += v[i]; }
#pragma unroll
  for (int m = 16; m; m >>= 1) s += __shfl_xor(s, m, 32);
  float mu = s * (1.0f / DD);
  float q = 0.0f;
#pragma unroll
  for (int i = 0; i < 12; ++i) { float d = v[i] - mu; q += d * d; }
#pragma unroll
  for (int m = 16; m; m >>= 1) q += __shfl_xor(q, m, 32);
  float rs = rsqrtf(q * (1.0f / DD) + 1e-5f);
#pragma unroll
  for (int i = 0; i < 12; ++i) {
    int c    = lane + i * 32;
    float o  = (v[i] - mu) * rs * g[c] + be[c];
    size_t p = (size_t)row * DD + c;
    hn[p]    = o;
    hnb[p]   = (__bf16)o;
  }
}

// ---------------------------------------------------------------------------
// Kernel 2: fused attention per (batch, head).  4 waves, head dim padded to 64.
// ---------------------------------------------------------------------------
__global__ __launch_bounds__(128) void attn_kernel(
    const __bf16* __restrict__ hnb, const __bf16* __restrict__ wqb,
    const float* __restrict__ bq, const __bf16* __restrict__ wkb,
    const float* __restrict__ bk, const __bf16* __restrict__ wvb,
    const float* __restrict__ bv, __bf16* __restrict__ attnb) {
  __shared__ __bf16 sA[64 * 64];   // hnorm K-chunk (TDM target)
  __shared__ __bf16 sQ[64 * 64];   // Q, padded 48->64
  __shared__ __bf16 sK[64 * 64];   // K, padded
  __shared__ __bf16 sVT[64 * 64];  // V TRANSPOSED: sVT[e][t]
  __shared__ __bf16 sP[64 * 64];   // softmax probs

  int b = blockIdx.x, h = blockIdx.y;
  int tid = threadIdx.x, w = tid >> 5, lane = tid & 31;
  int n16 = lane & 15, hi = (lane >> 4) & 1;

  for (int i = tid; i < 64 * 64; i += 128) {
    sQ[i] = (__bf16)0.0f; sK[i] = (__bf16)0.0f; sVT[i] = (__bf16)0.0f;
  }

  const __bf16* wqh = wqb + (size_t)h * DD * HS;
  const __bf16* wkh = wkb + (size_t)h * DD * HS;
  const __bf16* wvh = wvb + (size_t)h * DD * HS;

  // ---- QKV projection: [64x384] x [384x48] (3 N-tiles), 12 K-steps ----
  v8f accQ[3], accK[3], accV[3];
#pragma unroll
  for (int nt = 0; nt < 3; ++nt) { accQ[nt] = zero8(); accK[nt] = zero8(); accV[nt] = zero8(); }

  for (int kc = 0; kc < 6; ++kc) {
    __syncthreads();  // previous sA fully consumed
    if (w == 0) {
      tdm_load_2d_bf16((unsigned)(uintptr_t)&sA[0],
                       hnb + ((size_t)b * TT) * DD + kc * 64,
                       DD, TT, DD, 64, 64);
      __builtin_amdgcn_s_wait_tensorcnt(0);
    }
    __syncthreads();
#pragma unroll
    for (int ks = 0; ks < 2; ++ks) {
      v16bf a = load_a_lds(&sA[(16 * w) * 64 + ks * 32], 64, lane);
      int kt = kc * 2 + ks;  // fragment k-tile index (of 12)
#pragma unroll
      for (int nt = 0; nt < 3; ++nt) {
        accQ[nt] = wmma_bf16(a, load_b_packed(wqh, kt * 3 + nt, lane), accQ[nt]);
        accK[nt] = wmma_bf16(a, load_b_packed(wkh, kt * 3 + nt, lane), accK[nt]);
        accV[nt] = wmma_bf16(a, load_b_packed(wvh, kt * 3 + nt, lane), accV[nt]);
      }
    }
  }
  __syncthreads();
#pragma unroll
  for (int nt = 0; nt < 3; ++nt) {
    int col   = nt * 16 + n16;
    float biq = bq[h * HS + col], bik = bk[h * HS + col], biv = bv[h * HS + col];
    v8bf vv;
#pragma unroll
    for (int r = 0; r < 8; ++r) {
      int row = 16 * w + r + hi * 8;
      sQ[row * 64 + col] = (__bf16)(accQ[nt][r] + biq);
      sK[row * 64 + col] = (__bf16)(accK[nt][r] + bik);
      vv[r] = (__bf16)(accV[nt][r] + biv);
    }
    // transposed V: rows contiguous -> one 16B LDS store per lane
    *(v8bf*)&sVT[col * 64 + 16 * w + hi * 8] = vv;
  }
  __syncthreads();

  // ---- S = Q * K^T (zero-padded head dim -> exact), causal softmax ----
  v8f accS[4];
#pragma unroll
  for (int nt = 0; nt < 4; ++nt) accS[nt] = zero8();
#pragma unroll
  for (int ks = 0; ks < 2; ++ks) {
    v16bf a = load_a_lds(&sQ[(16 * w) * 64 + ks * 32], 64, lane);
#pragma unroll
    for (int nt = 0; nt < 4; ++nt) {
      v16bf f  = load_bT_lds(&sK[(nt * 16) * 64 + ks * 32], 64, lane);
      accS[nt] = wmma_bf16(a, f, accS[nt]);
    }
  }
  const float scale = 0.14433756729740643f;  // 1/sqrt(48)
#pragma unroll
  for (int r = 0; r < 8; ++r) {
    int row = 16 * w + r + hi * 8;
    float sv[4], mx = -1e30f;
#pragma unroll
    for (int nt = 0; nt < 4; ++nt) {
      int col = nt * 16 + n16;
      float s = accS[nt][r] * scale;
      sv[nt]  = (col <= row) ? s : -1e30f;
      mx      = fmaxf(mx, sv[nt]);
    }
#pragma unroll
    for (int m = 8; m; m >>= 1) mx = fmaxf(mx, __shfl_xor(mx, m, 32));
    float p[4], sum = 0.0f;
#pragma unroll
    for (int nt = 0; nt < 4; ++nt) { p[nt] = __expf(sv[nt] - mx); sum += p[nt]; }
#pragma unroll
    for (int m = 8; m; m >>= 1) sum += __shfl_xor(sum, m, 32);
    float inv = 1.0f / sum;
#pragma unroll
    for (int nt = 0; nt < 4; ++nt)
      sP[row * 64 + nt * 16 + n16] = (__bf16)(p[nt] * inv);
  }
  __syncthreads();

  // ---- O = P * V  (B from transposed V: contiguous per lane) ----
  v8f accO[3];
#pragma unroll
  for (int nt = 0; nt < 3; ++nt) accO[nt] = zero8();
#pragma unroll
  for (int ks = 0; ks < 2; ++ks) {
    v16bf a = load_a_lds(&sP[(16 * w) * 64 + ks * 32], 64, lane);
#pragma unroll
    for (int nt = 0; nt < 3; ++nt) {
      v16bf f  = load_bT_lds(&sVT[(nt * 16) * 64 + ks * 32], 64, lane);
      accO[nt] = wmma_bf16(a, f, accO[nt]);
    }
  }
#pragma unroll
  for (int nt = 0; nt < 3; ++nt) {
    int col = nt * 16 + n16;  // < 48
#pragma unroll
    for (int r = 0; r < 8; ++r) {
      int row = 16 * w + r + hi * 8;
      attnb[((size_t)b * TT + row) * DD + h * HS + col] = (__bf16)accO[nt][r];
    }
  }
}

// ---------------------------------------------------------------------------
// Kernel 3: y = attn*Wo + bo + hnorm ; h2 = LN2(y) (f32 + bf16 mirror).
// ---------------------------------------------------------------------------
__global__ __launch_bounds__(256) void proj_kernel(
    const __bf16* __restrict__ attnb, const __bf16* __restrict__ wob,
    const float* __restrict__ bo, const float* __restrict__ hn,
    const float* __restrict__ g2, const float* __restrict__ be2,
    float* __restrict__ h2, __bf16* __restrict__ h2b) {
  __shared__ __bf16 sA[64 * 64];
  __shared__ float sSum[2][64];
  __shared__ float sSq[2][64];

  int b = blockIdx.x, tid = threadIdx.x, w = tid >> 5, lane = tid & 31;
  int mt = w & 3, half = w >> 2;
  int n16 = lane & 15, hi = (lane >> 4) & 1;

  v8f acc[12];
#pragma unroll
  for (int nt = 0; nt < 12; ++nt) acc[nt] = zero8();

  for (int kc = 0; kc < 6; ++kc) {
    __syncthreads();
    if (w == 0) {
      tdm_load_2d_bf16((unsigned)(uintptr_t)&sA[0],
                       attnb + ((size_t)b * TT) * DD + kc * 64,
                       DD, TT, DD, 64, 64);
      __builtin_amdgcn_s_wait_tensorcnt(0);
    }
    __syncthreads();
#pragma unroll
    for (int ks = 0; ks < 2; ++ks) {
      v16bf a = load_a_lds(&sA[(16 * mt) * 64 + ks * 32], 64, lane);
      int kt = kc * 2 + ks;
#pragma unroll
      for (int nt = 0; nt < 12; ++nt) {
        int fragn = half * 12 + nt;  // of 24 N-tiles
        acc[nt] = wmma_bf16(a, load_b_packed(wob, kt * 24 + fragn, lane), acc[nt]);
      }
    }
  }

  // bias + residual, then LN2 stats (E[x], E[x^2]) across both halves via LDS
#pragma unroll
  for (int r = 0; r < 8; ++r) {
    int row = 16 * mt + r + hi * 8;
    float s = 0.0f, q = 0.0f;
#pragma unroll
    for (int nt = 0; nt < 12; ++nt) {
      int col    = (half * 12 + nt) * 16 + n16;
      float v    = acc[nt][r] + bo[col] + hn[((size_t)b * TT + row) * DD + col];
      acc[nt][r] = v;
      s += v; q += v * v;
    }
#pragma unroll
    for (int m = 8; m; m >>= 1) { s += __shfl_xor(s, m, 32); q += __shfl_xor(q, m, 32); }
    if (n16 == 0) { sSum[half][row] = s; sSq[half][row] = q; }
  }
  __syncthreads();
#pragma unroll
  for (int r = 0; r < 8; ++r) {
    int row   = 16 * mt + r + hi * 8;
    float tot = sSum[0][row] + sSum[1][row];
    float tq  = sSq[0][row] + sSq[1][row];
    float mu  = tot * (1.0f / DD);
    float var = tq * (1.0f / DD) - mu * mu;
    float rs  = rsqrtf(var + 1e-5f);
#pragma unroll
    for (int nt = 0; nt < 12; ++nt) {
      int col    = (half * 12 + nt) * 16 + n16;
      float o    = (acc[nt][r] - mu) * rs * g2[col] + be2[col];
      size_t idx = ((size_t)b * TT + row) * DD + col;
      h2[idx]  = o;
      h2b[idx] = (__bf16)o;
    }
  }
}

// ---------------------------------------------------------------------------
// Kernel 4: out = h2 + relu(h2*W1 + b1)*W2 + b2, fused per batch element.
// ---------------------------------------------------------------------------
__global__ __launch_bounds__(256) void ffn_kernel(
    const float* __restrict__ h2, const __bf16* __restrict__ h2b,
    const __bf16* __restrict__ w1b, const float* __restrict__ b1,
    const __bf16* __restrict__ w2b, const float* __restrict__ b2,
    float* __restrict__ out) {
  __shared__ __bf16 sH[64 * DD];  // 48 KB: full h2 row-block (TDM target)
  __shared__ __bf16 sU[64 * 64];  // 8 KB: ReLU chunk

  int b = blockIdx.x, tid = threadIdx.x, w = tid >> 5, lane = tid & 31;
  int mt = w & 3, half = w >> 2;
  int n16 = lane & 15, hi = (lane >> 4) & 1;

  if (w == 0) {
    tdm_load_2d_bf16((unsigned)(uintptr_t)&sH[0],
                     h2b + (size_t)b * TT * DD,
                     DD, TT, DD, DD, TT);  // 64 rows x 384 cols, contiguous
    __builtin_amdgcn_s_wait_tensorcnt(0);
  }
  __syncthreads();

  v8f acc[12];
#pragma unroll
  for (int nt = 0; nt < 12; ++nt) acc[nt] = zero8();

  for (int fc = 0; fc < 24; ++fc) {  // 24 chunks of 64 over F=1536
    v8f u[2];
    u[0] = zero8(); u[1] = zero8();
    int ntb = half * 2;
#pragma unroll
    for (int ks = 0; ks < 12; ++ks) {
      v16bf a = load_a_lds(&sH[(16 * mt) * DD + ks * 32], DD, lane);
#pragma unroll
      for (int j = 0; j < 2; ++j) {
        int fragn = fc * 4 + ntb + j;  // of 96 N-tiles
        u[j] = wmma_bf16(a, load_b_packed(w1b, ks * 96 + fragn, lane), u[j]);
      }
    }
    __syncthreads();  // previous sU fully consumed
#pragma unroll
    for (int j = 0; j < 2; ++j) {
      int colL   = (ntb + j) * 16 + n16;
      float bias = b1[fc * 64 + colL];
#pragma unroll
      for (int r = 0; r < 8; ++r) {
        int row = 16 * mt + r + hi * 8;
        sU[row * 64 + colL] = (__bf16)fmaxf(u[j][r] + bias, 0.0f);
      }
    }
    __syncthreads();
#pragma unroll
    for (int ks = 0; ks < 2; ++ks) {
      v16bf a = load_a_lds(&sU[(16 * mt) * 64 + ks * 32], 64, lane);
      int kt = fc * 2 + ks;  // of 48 K-tiles
#pragma unroll
      for (int nt = 0; nt < 12; ++nt) {
        int fragn = half * 12 + nt;  // of 24 N-tiles
        acc[nt] = wmma_bf16(a, load_b_packed(w2b, kt * 24 + fragn, lane), acc[nt]);
      }
    }
  }

#pragma unroll
  for (int nt = 0; nt < 12; ++nt) {
    int col    = (half * 12 + nt) * 16 + n16;
    float bias = b2[col];
#pragma unroll
    for (int r = 0; r < 8; ++r) {
      int row    = 16 * mt + r + hi * 8;
      size_t idx = ((size_t)b * TT + row) * DD + col;
      out[idx]   = h2[idx] + acc[nt][r] + bias;
    }
  }
}

// ---------------------------------------------------------------------------
extern "C" void kernel_launch(void* const* d_in, const int* in_sizes, int n_in,
                              void* d_out, int out_size, void* d_ws, size_t ws_size,
                              hipStream_t stream) {
  const float* x   = (const float*)d_in[0];
  const float* wq  = (const float*)d_in[1];
  const float* bq  = (const float*)d_in[2];
  const float* wk  = (const float*)d_in[3];
  const float* bk  = (const float*)d_in[4];
  const float* wv  = (const float*)d_in[5];
  const float* bv  = (const float*)d_in[6];
  const float* wo  = (const float*)d_in[7];
  const float* bo  = (const float*)d_in[8];
  const float* w1  = (const float*)d_in[9];
  const float* b1  = (const float*)d_in[10];
  const float* w2  = (const float*)d_in[11];
  const float* b2  = (const float*)d_in[12];
  const float* g1  = (const float*)d_in[13];
  const float* be1 = (const float*)d_in[14];
  const float* g2  = (const float*)d_in[15];
  const float* be2 = (const float*)d_in[16];
  float* out = (float*)d_out;

  const size_t NTOK  = (size_t)BATCH * TT;            // 131072 rows
  const size_t F32B  = NTOK * DD * sizeof(float);     // 201.3 MB
  const size_t BF16B = NTOK * DD * sizeof(short);     // 100.7 MB
  const size_t WQKV  = (size_t)NH * DD * HS;          // 147456 elems each
  const size_t WO    = (size_t)DD * DD;               // 147456
  const size_t W1N   = (size_t)DD * FF;               // 589824
  const size_t W2N   = (size_t)FF * DD;               // 589824

  char* ws = (char*)d_ws;
  size_t off = 0;
  float*  hn    = (float*)(ws + off);  off += F32B;
  __bf16* hnb   = (__bf16*)(ws + off); off += BF16B;
  __bf16* attnb = (__bf16*)(ws + off); off += BF16B;
  float*  h2    = (float*)(ws + off);  off += F32B;
  __bf16* h2b   = (__bf16*)(ws + off); off += BF16B;
  __bf16* wqb   = (__bf16*)(ws + off); off += WQKV * 2;
  __bf16* wkb   = (__bf16*)(ws + off); off += WQKV * 2;
  __bf16* wvb   = (__bf16*)(ws + off); off += WQKV * 2;
  __bf16* wob   = (__bf16*)(ws + off); off += WO * 2;
  __bf16* w1b   = (__bf16*)(ws + off); off += W1N * 2;
  __bf16* w2b   = (__bf16*)(ws + off); off += W2N * 2;

  (void)in_sizes; (void)n_in; (void)out_size; (void)ws_size;

  // 0) one-shot weight pack into WMMA-B fragment order (bf16)
  {
    // per-head QKV: K=384, N=48 -> 36 frags -> 1152 threads per head
    dim3 gq((1152 + 255) / 256, NH);
    packB_kernel<<<gq, 256, 0, stream>>>(wq, wqb, DD, HS);
    packB_kernel<<<gq, 256, 0, stream>>>(wk, wkb, DD, HS);
    packB_kernel<<<gq, 256, 0, stream>>>(wv, wvb, DD, HS);
    // Wo: K=384, N=384 -> 288 frags -> 9216 threads
    packB_kernel<<<dim3(36, 1), 256, 0, stream>>>(wo, wob, DD, DD);
    // W1: K=384, N=1536 -> 1152 frags -> 36864 threads
    packB_kernel<<<dim3(144, 1), 256, 0, stream>>>(w1, w1b, DD, FF);
    // W2: K=1536, N=384 -> 1152 frags
    packB_kernel<<<dim3(144, 1), 256, 0, stream>>>(w2, w2b, FF, DD);
  }

  // 1) LayerNorm1
  ln1_kernel<<<(int)(NTOK / 8), 256, 0, stream>>>(x, g1, be1, hn, hnb, (int)NTOK);
  // 2) fused causal attention per (batch, head)
  attn_kernel<<<dim3(BATCH, NH), 128, 0, stream>>>(hnb, wqb, bq, wkb, bk, wvb, bv, attnb);
  // 3) output projection + residual + LayerNorm2
  proj_kernel<<<BATCH, 256, 0, stream>>>(attnb, wob, bo, hn, g2, be2, h2, h2b);
  // 4) fused FFN + final residual
  ffn_kernel<<<BATCH, 256, 0, stream>>>(h2, h2b, w1b, b1, w2b, b2, out);
}
